// DecoderModel_49727131353956
// MI455X (gfx1250) — compile-verified, matrix operand order checked by reading, not run on hardware
//
#include <hip/hip_runtime.h>
#include <stdint.h>

typedef __attribute__((ext_vector_type(16))) _Float16 v16h;
typedef __attribute__((ext_vector_type(8)))  _Float16 v8h;
typedef __attribute__((ext_vector_type(8)))  float    v8f;
typedef __attribute__((ext_vector_type(4))) unsigned int u32x4;
typedef __attribute__((ext_vector_type(8))) int          i32x8;
typedef __attribute__((ext_vector_type(4))) int          i32x4;

#define TT    300
#define NB    64
#define LL    600
#define DD    512
#define KK    256
#define VV    256
#define VOCAB 1001
#define VOCP  1024
#define CHK   256   // A-chunk width (f16 elems) staged via TDM

// ---------------- device helpers ----------------

// Load one 16x32 f16 WMMA operand tile (A or B-as-W-rows).
// Lanes 0-15 -> rows 0-15 with K-base 0, lanes 16-31 -> rows 0-15 with K-base 8;
// each lane holds K = kb..kb+7 and kb+16..kb+23 (two 16B runs).
__device__ __forceinline__ v16h ld_tile(const _Float16* __restrict__ base, int ld, int lane) {
  const int r  = lane & 15;
  const int kb = (lane >> 4) << 3;
  const _Float16* p = base + (size_t)r * ld + kb;
  v8h lo = *(const v8h*)(p);
  v8h hi = *(const v8h*)(p + 16);
  v16h o;
#pragma unroll
  for (int i = 0; i < 8; ++i) { o[i] = lo[i]; o[i + 8] = hi[i]; }
  return o;
}

__device__ __forceinline__ float sigf(float x) { return 1.0f / (1.0f + __expf(-x)); }

// TDM: async-load a 2D f16 tile (rows x (d0_units*4 elems)) from global into LDS.
// D# built per CDNA5 ISA 8.3/8.4: data_size=3 (8B units), 2D tile, type=2.
__device__ __forceinline__ void tdm_load_2d(uint32_t lds_off, const void* gsrc,
                                            uint32_t tile_d0_u,   // row length, 8B units
                                            uint32_t rows,        // tile/tensor dim1
                                            uint32_t stride_u)    // row stride, 8B units
{
  const uint64_t ga = (uint64_t)(uintptr_t)gsrc;
  u32x4 g0;
  g0[0] = 1u;                                          // count=1 (valid user D#)
  g0[1] = lds_off;                                     // lds_addr (bytes)
  g0[2] = (uint32_t)ga;                                // global_addr[31:0]
  g0[3] = ((uint32_t)(ga >> 32) & 0x01FFFFFFu) | 0x80000000u; // ga[56:32] | type=2

  const uint32_t td0 = stride_u;                       // tensor_dim0 >= tile_dim0 (no OOB clip)
  i32x8 g1;
  g1[0] = (int)(3u << 16);                             // wg_mask=0, data_size=3 (8B)
  g1[1] = (int)((td0 & 0xFFFFu) << 16);                // tensor_dim0[15:0]
  g1[2] = (int)(((td0 >> 16) & 0xFFFFu) | ((rows & 0xFFFFu) << 16));   // td0[31:16] | tensor_dim1[15:0]
  g1[3] = (int)((tile_d0_u & 0xFFFFu) << 16);          // tensor_dim1[31:16]=0 | tile_dim0
  g1[4] = (int)(rows & 0xFFFFu);                       // tile_dim1 | tile_dim2=0
  g1[5] = (int)stride_u;                               // tensor_dim0_stride[31:0]
  g1[6] = 0;                                           // stride[47:32]=0 | d1stride[15:0]=0
  g1[7] = 0;
  i32x4 z4 = {};
#if __clang_major__ >= 23
  i32x8 z8 = {};
  __builtin_amdgcn_tensor_load_to_lds(g0, g1, z4, z4, z8, 0);
#else
  __builtin_amdgcn_tensor_load_to_lds(g0, g1, z4, z4, 0);
#endif
}

// ---------------- fused LSTM cell: TDM-staged A + WMMA gates GEMM + pointwise ----------------
// gates(64 x 2048) = A(64 x Kd) @ W(2048 x Kd)^T + bias ; gate order i,f,g,o (chunks of 512)
// grid.x = 8 (64 d-columns per block), block = 256 threads (8 waves).
// A is staged into LDS in double-buffered 64x256 f16 chunks by the Tensor Data Mover;
// B (weights) streams from L2 under the WMMAs. Each wave owns all 4 gates for its
// (16 rows x 32 cols) patch -> pointwise LSTM update fully in-register.
__global__ void __launch_bounds__(256)
k_lstm(const _Float16* __restrict__ A, int Kd,
       const _Float16* __restrict__ W,
       const float* __restrict__ bias,
       float* __restrict__ cst,
       _Float16* __restrict__ hdst, int ldh)
{
  __shared__ _Float16 sA[2][NB * CHK];   // 2 x 32KB

  const int lane = threadIdx.x & 31;
  const int wv   = threadIdx.x >> 5;
  const int rs   = wv & 3;
  const int ch   = wv >> 2;
  const int dblk = blockIdx.x * 64 + ch * 32;

  v8f acc[4][2] = {};
  const int nc = Kd / CHK;

  if (wv == 0)
    tdm_load_2d((uint32_t)(uintptr_t)&sA[0][0], A, CHK / 4, NB, (uint32_t)(Kd / 4));

  for (int c = 0; c < nc; ++c) {
    if (wv == 0) {
      if (c + 1 < nc) {
        tdm_load_2d((uint32_t)(uintptr_t)&sA[(c + 1) & 1][0], A + (size_t)(c + 1) * CHK,
                    CHK / 4, NB, (uint32_t)(Kd / 4));
        __builtin_amdgcn_s_wait_tensorcnt(1);   // chunk c resident (TDM in-order per wave)
      } else {
        __builtin_amdgcn_s_wait_tensorcnt(0);
      }
    }
    __syncthreads();

    const _Float16* Ab = &sA[c & 1][rs * 16 * CHK];
    const int kbase = c * CHK;
    for (int kk = 0; kk < CHK; kk += 32) {
      v16h at = ld_tile(Ab + kk, CHK, lane);
#pragma unroll
      for (int g = 0; g < 4; ++g) {
#pragma unroll
        for (int cc = 0; cc < 2; ++cc) {
          const int nrow = g * DD + dblk + cc * 16;
          v16h bt = ld_tile(W + (size_t)nrow * Kd + kbase + kk, Kd, lane);
          acc[g][cc] = __builtin_amdgcn_wmma_f32_16x16x32_f16(
              false, at, false, bt, (short)0, acc[g][cc], false, false);
        }
      }
      __builtin_prefetch(W + (size_t)dblk * Kd + kbase + kk + 256, 0, 1);
    }
    __syncthreads();   // all waves done with buffer (c&1) before TDM overwrites it
  }

  // C/D layout: lane&15 = column within tile; rows = ((lane>>4)<<3)+j
#pragma unroll
  for (int c = 0; c < 2; ++c) {
    const int d = dblk + c * 16 + (lane & 15);
#pragma unroll
    for (int j = 0; j < 8; ++j) {
      const int row = rs * 16 + ((lane >> 4) << 3) + j;
      const float gi = acc[0][c][j] + bias[d];
      const float gf = acc[1][c][j] + bias[DD + d];
      const float gg = acc[2][c][j] + bias[2 * DD + d];
      const float go = acc[3][c][j] + bias[3 * DD + d];
      const float cp = cst[(size_t)row * DD + d];
      const float cn = sigf(gf) * cp + sigf(gi) * tanhf(gg);
      const float hn = sigf(go) * tanhf(cn);
      cst[(size_t)row * DD + d]  = cn;
      hdst[(size_t)row * ldh + d] = (_Float16)hn;
    }
  }
}

// ---------------- generic WMMA GEMM + bias (+ optional leaky-relu), f32 and/or f16 out ----
// C(64 x N) = act(A(64 x Kd, stride lda) @ W(N x Kd)^T + bias); grid.x = N/128, block = 256.
__global__ void __launch_bounds__(256)
k_gemm(const _Float16* __restrict__ A, int lda,
       const _Float16* __restrict__ W, int Kd,
       const float* __restrict__ bias,
       float* __restrict__ Cf, int ldc,
       _Float16* __restrict__ Ch, int ldch,
       int act)
{
  const int lane = threadIdx.x & 31;
  const int wv   = threadIdx.x >> 5;
  const int rs   = wv & 3;
  const int cg   = wv >> 2;
  const int nb0  = blockIdx.x * 128 + cg * 64;

  v8f acc[4] = {};
  const _Float16* Ab = A + (size_t)rs * 16 * lda;

  for (int k0 = 0; k0 < Kd; k0 += 32) {
    v16h at = ld_tile(Ab + k0, lda, lane);
#pragma unroll
    for (int c = 0; c < 4; ++c) {
      const int nrow = nb0 + c * 16;
      v16h bt = ld_tile(W + (size_t)nrow * Kd + k0, Kd, lane);
      acc[c] = __builtin_amdgcn_wmma_f32_16x16x32_f16(
          false, at, false, bt, (short)0, acc[c], false, false);
    }
    __builtin_prefetch(W + (size_t)nb0 * Kd + k0 + 256, 0, 1);
  }

#pragma unroll
  for (int c = 0; c < 4; ++c) {
    const int n = nb0 + c * 16 + (lane & 15);
#pragma unroll
    for (int j = 0; j < 8; ++j) {
      const int row = rs * 16 + ((lane >> 4) << 3) + j;
      float v = acc[c][j] + bias[n];
      if (act == 1) v = (v > 0.0f) ? v : 0.01f * v;
      if (Cf) Cf[(size_t)row * ldc + n] = v;
      if (Ch) Ch[(size_t)row * ldch + n] = (_Float16)v;
    }
  }
}

// ---------------- per-sample masked softmax attention ----------------
__global__ void __launch_bounds__(256)
k_attn(const float* __restrict__ keys, const float* __restrict__ vals,
       const int* __restrict__ ulen, const float* __restrict__ q,
       float* __restrict__ attn_out,
       _Float16* __restrict__ ctx0, int ld0,
       _Float16* __restrict__ ctx1, int ld1)
{
  const int n = blockIdx.x, tid = threadIdx.x;
  __shared__ float sq[KK];
  __shared__ float sa[LL];
  __shared__ float red[256];

  if (tid < KK) sq[tid] = q[(size_t)n * KK + tid];
  __syncthreads();

  const int ul = ulen[n];
  for (int l = tid; l < LL; l += 256) {
    const float* kr = keys + ((size_t)l * NB + n) * KK;
    float dot = 0.0f;
#pragma unroll 4
    for (int k = 0; k < KK; ++k) dot += kr[k] * sq[k];
    const float m = (l < ul) ? 1.0f : 0.0f;
    sa[l] = dot * m - (1.0f - m) * 10000.0f;
  }
  __syncthreads();

  float lm = -3.0e38f;
  for (int l = tid; l < LL; l += 256) lm = fmaxf(lm, sa[l]);
  red[tid] = lm; __syncthreads();
  for (int s = 128; s > 0; s >>= 1) { if (tid < s) red[tid] = fmaxf(red[tid], red[tid + s]); __syncthreads(); }
  const float mx = red[0]; __syncthreads();

  float ls = 0.0f;
  for (int l = tid; l < LL; l += 256) {
    const float m = (l < ul) ? 1.0f : 0.0f;
    const float ev = __expf(sa[l] - mx) * m;
    sa[l] = ev; ls += ev;
  }
  __syncthreads();
  red[tid] = ls; __syncthreads();
  for (int s = 128; s > 0; s >>= 1) { if (tid < s) red[tid] += red[tid + s]; __syncthreads(); }
  const float inv = 1.0f / red[0]; __syncthreads();

  for (int l = tid; l < LL; l += 256) {
    const float a = sa[l] * inv;
    sa[l] = a;
    attn_out[(size_t)n * LL + l] = a;
  }
  __syncthreads();

  for (int v = tid; v < VV; v += 256) {
    float acc = 0.0f;
    for (int l = 0; l < LL; ++l) acc += sa[l] * vals[((size_t)l * NB + n) * VV + v];
    const _Float16 c16 = (_Float16)acc;
    ctx0[(size_t)n * ld0 + v] = c16;
    ctx1[(size_t)n * ld1 + v] = c16;
  }
}

// ---------------- gumbel argmax + logits copy-out ----------------
__global__ void __launch_bounds__(256)
k_sample(const float* __restrict__ logits, float* __restrict__ outl,
         float* __restrict__ outg, int t)
{
  const int n = blockIdx.x, tid = threadIdx.x;
  __shared__ float bv[256];
  __shared__ int   bi[256];
  float bestv = -3.0e38f; int besti = 0;
  for (int v = tid; v < VOCAB; v += 256) {
    const float lg = logits[(size_t)n * VOCP + v];
    outl[(size_t)n * VOCAB + v] = lg;
    uint32_t h = (uint32_t)t * 1315423911u ^ (uint32_t)n * 2654435761u ^ (uint32_t)v * 0x9E3779B9u;
    h ^= h >> 16; h *= 0x7feb352dU; h ^= h >> 15; h *= 0x846ca68bU; h ^= h >> 16;
    const float u = (h >> 8) * (1.0f / 16777216.0f);
    const float g = -__logf(1e-10f - __logf(u + 1e-10f));
    const float val = lg + g;
    if (val > bestv) { bestv = val; besti = v; }
  }
  bv[tid] = bestv; bi[tid] = besti; __syncthreads();
  for (int s = 128; s > 0; s >>= 1) {
    if (tid < s && bv[tid + s] > bv[tid]) { bv[tid] = bv[tid + s]; bi[tid] = bi[tid + s]; }
    __syncthreads();
  }
  if (tid == 0) outg[n] = (float)bi[0];
}

// ---------------- small utility kernels ----------------
__global__ void k_embed(const int* __restrict__ inputs, int t,
                        const _Float16* __restrict__ E16, _Float16* __restrict__ x1)
{
  const int n = blockIdx.x;
  int tok = inputs[(size_t)t * NB + n];
  tok = (tok < 0) ? 0 : (tok >= VOCAB ? VOCAB - 1 : tok);
  for (int c = threadIdx.x; c < DD; c += 256)
    x1[(size_t)n * 1280 + c] = E16[(size_t)tok * DD + c];
}

__global__ void k_cpy16(const _Float16* __restrict__ src, int lds,
                        _Float16* __restrict__ dst, int ldd)
{
  const int i = blockIdx.x * blockDim.x + threadIdx.x;
  if (i >= NB * DD) return;
  const int n = i / DD, d = i % DD;
  dst[(size_t)n * ldd + d] = src[(size_t)n * lds + d];
}

__global__ void k_concat_cvt(const float* __restrict__ a, int ca,
                             const float* __restrict__ b, int cb,
                             _Float16* __restrict__ d, int total)
{
  const int i = blockIdx.x * blockDim.x + threadIdx.x;
  if (i >= total) return;
  const int cab = ca + cb, row = i / cab, col = i % cab;
  const float v = (col < ca) ? a[(size_t)row * ca + col] : b[(size_t)row * cb + (col - ca)];
  d[i] = (_Float16)v;
}

__global__ void k_cvt(const float* __restrict__ s, _Float16* __restrict__ d, int n)
{
  const int i = blockIdx.x * blockDim.x + threadIdx.x;
  if (i < n) d[i] = (_Float16)s[i];
}

__global__ void k_cvtE(const float* __restrict__ E, _Float16* __restrict__ d)
{
  const int i = blockIdx.x * blockDim.x + threadIdx.x;
  if (i >= VOCP * DD) return;
  d[i] = (i / DD < VOCAB) ? (_Float16)E[i] : (_Float16)0.0f;
}

__global__ void k_add2(const float* __restrict__ a, const float* __restrict__ b,
                       float* __restrict__ d, int n)
{
  const int i = blockIdx.x * blockDim.x + threadIdx.x;
  if (i < n) d[i] = a[i] + b[i];
}

__global__ void k_padb2(const float* __restrict__ b2, float* __restrict__ d)
{
  const int i = blockIdx.x * blockDim.x + threadIdx.x;
  if (i < VOCP) d[i] = (i < VOCAB) ? b2[i] : -1.0e30f;
}

__global__ void k_bcast32(const float* __restrict__ src, float* __restrict__ dst)
{
  const int i = blockIdx.x * blockDim.x + threadIdx.x;
  if (i < NB * DD) dst[i] = src[i % DD];
}

__global__ void k_bcast16(const float* __restrict__ src, _Float16* __restrict__ dst, int ld)
{
  const int i = blockIdx.x * blockDim.x + threadIdx.x;
  if (i >= NB * DD) return;
  dst[(size_t)(i / DD) * ld + (i % DD)] = (_Float16)src[i % DD];
}

// ---------------- host launcher ----------------
extern "C" void kernel_launch(void* const* d_in, const int* in_sizes, int n_in,
                              void* d_out, int out_size, void* d_ws, size_t ws_size,
                              hipStream_t stream) {
  (void)in_sizes; (void)n_in; (void)out_size; (void)ws_size;
  const int*   inputs = (const int*)  d_in[0];
  const float* keys   = (const float*)d_in[2];
  const float* vals   = (const float*)d_in[3];
  const int*   ulen   = (const int*)  d_in[4];
  const float* E      = (const float*)d_in[5];
  const float* Wih1 = (const float*)d_in[6],  * Whh1 = (const float*)d_in[7];
  const float* bih1 = (const float*)d_in[8],  * bhh1 = (const float*)d_in[9];
  const float* h01  = (const float*)d_in[10], * c01  = (const float*)d_in[11];
  const float* Wih2 = (const float*)d_in[12], * Whh2 = (const float*)d_in[13];
  const float* bih2 = (const float*)d_in[14], * bhh2 = (const float*)d_in[15];
  const float* h02  = (const float*)d_in[16], * c02  = (const float*)d_in[17];
  const float* Wih3 = (const float*)d_in[18], * Whh3 = (const float*)d_in[19];
  const float* bih3 = (const float*)d_in[20], * bhh3 = (const float*)d_in[21];
  const float* h03  = (const float*)d_in[22], * c03  = (const float*)d_in[23];
  const float* Wq   = (const float*)d_in[24], * bq   = (const float*)d_in[25];
  const float* W1   = (const float*)d_in[26], * b1   = (const float*)d_in[27];
  const float* b2   = (const float*)d_in[28];

  char* ws = (char*)d_ws;
  size_t off = 0;
  auto alloc = [&](size_t bytes) -> void* {
    void* p = ws + off;
    off = (off + bytes + 255) & ~(size_t)255;
    return p;
  };
  _Float16* Wg1   = (_Float16*)alloc((size_t)2048 * 1280 * 2);
  _Float16* Wg2   = (_Float16*)alloc((size_t)2048 * 1024 * 2);
  _Float16* Wg3   = (_Float16*)alloc((size_t)2048 * 1024 * 2);
  _Float16* Wq16  = (_Float16*)alloc((size_t)256 * 512 * 2);
  _Float16* W116  = (_Float16*)alloc((size_t)512 * 768 * 2);
  _Float16* E16   = (_Float16*)alloc((size_t)VOCP * DD * 2);
  _Float16* xc1   = (_Float16*)alloc((size_t)NB * 1280 * 2);   // [E | ctx | h1]
  _Float16* xc2   = (_Float16*)alloc((size_t)NB * 1024 * 2);   // [h1 | h2]
  _Float16* xc3   = (_Float16*)alloc((size_t)NB * 1024 * 2);   // [h2 | h3]
  _Float16* hcat  = (_Float16*)alloc((size_t)NB * 768 * 2);    // [h3 | ctx]
  _Float16* hid16 = (_Float16*)alloc((size_t)NB * DD * 2);
  float* bias1 = (float*)alloc(2048 * 4);
  float* bias2 = (float*)alloc(2048 * 4);
  float* bias3 = (float*)alloc(2048 * 4);
  float* b2p   = (float*)alloc(VOCP * 4);
  float* c1    = (float*)alloc((size_t)NB * DD * 4);
  float* c2    = (float*)alloc((size_t)NB * DD * 4);
  float* c3    = (float*)alloc((size_t)NB * DD * 4);
  float* qbuf  = (float*)alloc((size_t)NB * KK * 4);
  float* lgbuf = (float*)alloc((size_t)NB * VOCP * 4);
  float* adum  = (float*)alloc((size_t)NB * LL * 4);

  float* outL = (float*)d_out;
  float* outA = outL + (size_t)TT * NB * VOCAB;
  float* outG = outA + (size_t)TT * NB * LL;

  const int BT = 256;
  auto nb = [](size_t n) { return (unsigned)((n + 255) / 256); };

  // ---- one-time prep (re-run every call: deterministic) ----
  k_concat_cvt<<<nb((size_t)2048 * 1280), BT, 0, stream>>>(Wih1, 768, Whh1, 512, Wg1, 2048 * 1280);
  k_concat_cvt<<<nb((size_t)2048 * 1024), BT, 0, stream>>>(Wih2, 512, Whh2, 512, Wg2, 2048 * 1024);
  k_concat_cvt<<<nb((size_t)2048 * 1024), BT, 0, stream>>>(Wih3, 512, Whh3, 512, Wg3, 2048 * 1024);
  k_cvt<<<nb((size_t)256 * 512), BT, 0, stream>>>(Wq, Wq16, 256 * 512);
  k_cvt<<<nb((size_t)512 * 768), BT, 0, stream>>>(W1, W116, 512 * 768);
  k_cvtE<<<nb((size_t)VOCP * DD), BT, 0, stream>>>(E, E16);
  k_add2<<<nb(2048), BT, 0, stream>>>(bih1, bhh1, bias1, 2048);
  k_add2<<<nb(2048), BT, 0, stream>>>(bih2, bhh2, bias2, 2048);
  k_add2<<<nb(2048), BT, 0, stream>>>(bih3, bhh3, bias3, 2048);
  k_padb2<<<nb(VOCP), BT, 0, stream>>>(b2, b2p);
  k_bcast32<<<nb((size_t)NB * DD), BT, 0, stream>>>(c01, c1);
  k_bcast32<<<nb((size_t)NB * DD), BT, 0, stream>>>(c02, c2);
  k_bcast32<<<nb((size_t)NB * DD), BT, 0, stream>>>(c03, c3);
  k_bcast16<<<nb((size_t)NB * DD), BT, 0, stream>>>(h01, xc1 + 768, 1280);
  k_bcast16<<<nb((size_t)NB * DD), BT, 0, stream>>>(h02, xc2 + 512, 1024);
  k_bcast16<<<nb((size_t)NB * DD), BT, 0, stream>>>(h03, xc3 + 512, 1024);

  // ---- initial context: q = h3(=h03) @ Wq^T + bq ; attn -> ctx slices ----
  k_gemm<<<2, BT, 0, stream>>>(xc3 + 512, 1024, Wq16, 512, bq, qbuf, KK, nullptr, 0, 0);
  k_attn<<<NB, BT, 0, stream>>>(keys, vals, ulen, qbuf, adum, xc1 + 512, 1280, hcat + 512, 768);

  // ---- 300 sequential decode steps ----
  for (int t = 0; t < TT; ++t) {
    k_embed<<<NB, BT, 0, stream>>>(inputs, t, E16, xc1);

    // LSTM1: reads xc1, writes h1 -> xc2[:,0:512]; then refresh xc1 h-slice for t+1
    k_lstm<<<8, BT, 0, stream>>>(xc1, 1280, Wg1, bias1, c1, xc2, 1024);
    k_cpy16<<<nb((size_t)NB * DD), BT, 0, stream>>>(xc2, 1024, xc1 + 768, 1280);

    // LSTM2: reads xc2, writes h2 -> xc3[:,0:512]; refresh xc2 h2-slice
    k_lstm<<<8, BT, 0, stream>>>(xc2, 1024, Wg2, bias2, c2, xc3, 1024);
    k_cpy16<<<nb((size_t)NB * DD), BT, 0, stream>>>(xc3, 1024, xc2 + 512, 1024);

    // LSTM3: reads xc3, writes h3 -> hcat[:,0:512]; refresh xc3 h3-slice
    k_lstm<<<8, BT, 0, stream>>>(xc3, 1024, Wg3, bias3, c3, hcat, 768);
    k_cpy16<<<nb((size_t)NB * DD), BT, 0, stream>>>(hcat, 768, xc3 + 512, 1024);

    // q = h3 @ Wq^T + bq  (A = hcat[:,0:512], lda=768)
    k_gemm<<<2, BT, 0, stream>>>(hcat, 768, Wq16, 512, bq, qbuf, KK, nullptr, 0, 0);

    // attention -> a (output), ctx -> xc1[:,512:768] (next step) and hcat[:,512:768] (this step)
    k_attn<<<NB, BT, 0, stream>>>(keys, vals, ulen, qbuf,
                                  outA + (size_t)t * NB * LL,
                                  xc1 + 512, 1280, hcat + 512, 768);

    // hid = leaky_relu(hcat @ W1^T + b1) -> f16
    k_gemm<<<4, BT, 0, stream>>>(hcat, 768, W116, 768, b1, nullptr, 0, hid16, DD, 1);

    // logits = hid @ E^T + b2 (tied, N padded to 1024)
    k_gemm<<<8, BT, 0, stream>>>(hid16, DD, E16, DD, b2p, lgbuf, VOCP, nullptr, 0, 0);

    // gumbel argmax + copy logits to output
    k_sample<<<NB, BT, 0, stream>>>(lgbuf, outL + (size_t)t * NB * VOCAB,
                                    outG + (size_t)t * NB, t);
  }
}
